// VGG_Cifar10_34557306863995
// MI455X (gfx1250) — 8-bit matrix operand order repaired and checked by reading, compiled, not run on hardware
//
#include <hip/hip_runtime.h>
#include <hip/hip_bf16.h>
#include <stdint.h>


typedef __attribute__((ext_vector_type(8))) int v8i;

__device__ __forceinline__ int imax4(int a, int b, int c, int d) {
  int x = a > b ? a : b; int y = c > d ? c : d; return x > y ? x : y;
}
__device__ __forceinline__ int8_t fsign8(float f) {
  return f > 0.f ? (int8_t)1 : (f < 0.f ? (int8_t)-1 : (int8_t)0);
}

__device__ __forceinline__ v8i load_bfrag(const int8_t* __restrict__ pB,
                                          int ntile, int numK, int kc, int lane) {
  const uint4* bp = (const uint4*)(pB + ((((long)ntile * numK + kc) * 32 + lane) << 5));
  uint4 b0 = bp[0], b1 = bp[1];
  v8i b;
  b[0] = (int)b0.x; b[1] = (int)b0.y; b[2] = (int)b0.z; b[3] = (int)b0.w;
  b[4] = (int)b1.x; b[5] = (int)b1.y; b[6] = (int)b1.z; b[7] = (int)b1.w;
  return b;
}


__global__ void sign8_kernel(const float* __restrict__ in, int8_t* __restrict__ out, long n) {
  long i = (long)blockIdx.x * 256 + threadIdx.x;
  if (i < n) out[i] = fsign8(in[i]);
}

__global__ void pack_conv_w(const float* __restrict__ w, int8_t* __restrict__ pB,
                            int Cin, int Cout, long total) {
  long idx = (long)blockIdx.x * 256 + threadIdx.x;
  if (idx >= total) return;
  int byte = idx & 3;       long t = idx >> 2;
  int v    = (int)(t & 7);  t >>= 3;
  int lane = (int)(t & 31); t >>= 5;
  int numK = (9 * Cin) >> 6;
  int kc    = (int)(t % numK);
  int ntile = (int)(t / numK);
  int N = ntile * 16 + (lane & 15);
  int K = kc * 64 + (v >> 1) * 16 + (lane >> 4) * 8 + (v & 1) * 4 + byte;
  int ci = K % Cin, tap = K / Cin;
  float wv = w[((long)N * Cin + ci) * 9 + tap];
  pB[idx] = fsign8(wv);
}

__global__ void pack_fc_w(const float* __restrict__ w, int8_t* __restrict__ pB,
                          int K, int NoutReal, int permC, long total) {
  long idx = (long)blockIdx.x * 256 + threadIdx.x;
  if (idx >= total) return;
  int byte = idx & 3;       long t = idx >> 2;
  int v    = (int)(t & 7);  t >>= 3;
  int lane = (int)(t & 31); t >>= 5;
  int numK = K >> 6;
  int kc    = (int)(t % numK);
  int ntile = (int)(t / numK);
  int N = ntile * 16 + (lane & 15);
  int Kidx = kc * 64 + (v >> 1) * 16 + (lane >> 4) * 8 + (v & 1) * 4 + byte;
  int8_t r = 0;
  if (N < NoutReal) {
    int ksrc = Kidx;
    if (permC) { int c = Kidx % permC; int sp = Kidx / permC; ksrc = c * (K / permC) + sp; }
    r = fsign8(w[(long)N * K + ksrc]);
  }
  pB[idx] = r;
}

__global__ void bn_prep(const float* __restrict__ cb, const float* __restrict__ g,
                        const float* __restrict__ b, const float* __restrict__ m,
                        const float* __restrict__ v, float* __restrict__ scale,
                        float* __restrict__ shift, int C, int affine) {
  int i = blockIdx.x * 256 + threadIdx.x;
  if (i >= C) return;
  if (affine) {
    float s = g[i] * rsqrtf(v[i] + 1e-5f);
    scale[i] = s;
    shift[i] = (cb[i] - m[i]) * s + b[i];
  } else {
    float s = rsqrtf(v[i] + 1e-5f);
    scale[i] = s;
    shift[i] = (cb[i] - m[i]) * s;
  }
}


__global__ void conv1_kernel(const float* __restrict__ x, const int8_t* __restrict__ w1s,
                             const float* __restrict__ scale, const float* __restrict__ shift,
                             int8_t* __restrict__ a1) {
  long idx = (long)blockIdx.x * 256 + threadIdx.x;
  const long total = 256L * 32 * 32 * 384;
  if (idx >= total) return;
  int co = (int)(idx % 384); long t = idx / 384;
  int xx = (int)(t & 31); t >>= 5;
  int yy = (int)(t & 31); int n = (int)(t >> 5);
  float acc = 0.f;
  for (int ci = 0; ci < 3; ++ci)
    for (int ky = 0; ky < 3; ++ky) {
      int iy = yy + ky - 1; if ((unsigned)iy >= 32u) continue;
      for (int kx = 0; kx < 3; ++kx) {
        int ix = xx + kx - 1; if ((unsigned)ix >= 32u) continue;
        float xv = x[(((long)n * 3 + ci) * 32 + iy) * 32 + ix];
        acc += xv * (float)w1s[((co * 3 + ci) * 3 + ky) * 3 + kx];
      }
    }
  float f = acc * scale[co] + shift[co];
  a1[idx] = fsign8(f);
}


template <int POOL, int HH>
__global__ void __launch_bounds__(256)
bconv_wmma(const int8_t* __restrict__ actIn,
           const int8_t* __restrict__ pB,
           const float* __restrict__ scale, const float* __restrict__ shift,
           int8_t* __restrict__ actOut,
           int Cin, int Cout, int totalWaves) {
  constexpr int NT = 4;
  constexpr int H = HH, W = HH;
  constexpr int Ho = POOL ? HH / 2 : HH;
  constexpr int Wo = Ho;
  constexpr int mPer = POOL ? 4 : 16;
  constexpr int Mtiles = (256 * Ho * Wo) / mPer;
  __shared__ unsigned ldsA[8][16 * 16];

  const int lane = threadIdx.x & 31;
  const int wv   = threadIdx.x >> 5;
  const int gw   = blockIdx.x * 8 + wv;
  if (gw >= totalWaves) return;

  const int mt     = gw & (Mtiles - 1);
  const int ntBase = (gw / Mtiles) * NT;
  const int ccN    = Cin >> 6;
  const int numK   = 9 * ccN;

  const int mrow = lane & 15, half = lane >> 4;
  int fn, fy, fx;
  if (POOL) {
    int po = mt * 4 + (mrow >> 2);
    fn = po / (Ho * Wo); int r = po & (Ho * Wo - 1);
    int py = r / Wo, px = r & (Wo - 1);
    fy = 2 * py + ((mrow >> 1) & 1);
    fx = 2 * px + (mrow & 1);
  } else {
    int pp = mt * 16 + mrow;
    fn = pp / (H * W); int r = pp & (H * W - 1);
    fy = r / W; fx = r & (W - 1);
  }

  v8i acc[NT] = {};
  unsigned* myLds = &ldsA[wv][0];

  for (int tap = 0; tap < 9; ++tap) {
    int iy = fy + tap / 3 - 1;
    int ix = fx + tap % 3 - 1;
    bool inR = (iy >= 0) & (iy < H) & (ix >= 0) & (ix < W);
    const long rowOff = (((long)fn * H + iy) * W + ix) * Cin + half * 32;
    for (int cc = 0; cc < ccN; ++cc) {
      uint4 t0 = make_uint4(0, 0, 0, 0), t1 = make_uint4(0, 0, 0, 0);
      if (inR) {
        const uint4* s = (const uint4*)(actIn + rowOff + cc * 64);
        t0 = s[0]; t1 = s[1];
      }
      uint4* d = (uint4*)&myLds[mrow * 16 + half * 8];
      d[0] = t0; d[1] = t1;

      const unsigned* rw = &myLds[(lane & 15) * 16 + (half << 1)];
      v8i a;
      a[0] = (int)rw[0];  a[1] = (int)rw[1];
      a[2] = (int)rw[4];  a[3] = (int)rw[5];
      a[4] = (int)rw[8];  a[5] = (int)rw[9];
      a[6] = (int)rw[12]; a[7] = (int)rw[13];

      int kc = tap * ccN + cc;
      if (kc + 1 < numK)
        __builtin_prefetch(pB + ((((long)ntBase * numK + kc + 1) * 32 + lane) << 5), 0, 1);
#pragma unroll
      for (int nt = 0; nt < NT; ++nt) {
        v8i b = load_bfrag(pB, ntBase + nt, numK, kc, lane);
        acc[nt] = __builtin_amdgcn_wmma_i32_16x16x64_iu8(
            true, a, true, b, acc[nt], false, false);
      }
    }
  }

#pragma unroll
  for (int nt = 0; nt < NT; ++nt) {
    int ch = (ntBase + nt) * 16 + (lane & 15);
    float sc = scale[ch], sf = shift[ch];
    if (POOL) {
      int p02 = imax4(acc[nt][0], acc[nt][1], acc[nt][2], acc[nt][3]);
      int p13 = imax4(acc[nt][4], acc[nt][5], acc[nt][6], acc[nt][7]);
      int pv[2] = {p02, p13};
#pragma unroll
      for (int k = 0; k < 2; ++k) {
        int po = mt * 4 + half * 2 + k;
        int n = po / (Ho * Wo); int r = po & (Ho * Wo - 1);
        int py = r / Wo, px = r & (Wo - 1);
        actOut[(((long)n * Ho + py) * Wo + px) * Cout + ch] =
            fsign8((float)pv[k] * sc + sf);
      }
    } else {
#pragma unroll
      for (int rr = 0; rr < 8; ++rr) {
        int pp = mt * 16 + rr + half * 8;
        int n = pp / (H * W); int r = pp & (H * W - 1);
        int y = r / W, x = r & (W - 1);
        actOut[(((long)n * H + y) * W + x) * Cout + ch] =
            fsign8((float)acc[nt][rr] * sc + sf);
      }
    }
  }
}


template <int NT, int SOUT>
__global__ void __launch_bounds__(256)
bfc_wmma(const int8_t* __restrict__ actIn,
         const int8_t* __restrict__ pB,
         const float* __restrict__ scale, const float* __restrict__ shift,
         int8_t* __restrict__ outS, float* __restrict__ outF,
         int K, int NoutReal, int totalWaves) {
  const int lane = threadIdx.x & 31;
  const int gw   = blockIdx.x * 8 + (threadIdx.x >> 5);
  if (gw >= totalWaves) return;
  const int mt     = gw & 15;
  const int ntBase = (gw >> 4) * NT;
  const int numK   = K >> 6;
  const int half   = lane >> 4;
  const int8_t* arow = actIn + (long)(mt * 16 + (lane & 15)) * K + half * 8;

  v8i acc[NT] = {};
  for (int kc = 0; kc < numK; ++kc) {
    const unsigned* rp = (const unsigned*)(arow + kc * 64);
    v8i a;
    a[0] = (int)rp[0];  a[1] = (int)rp[1];
    a[2] = (int)rp[4];  a[3] = (int)rp[5];
    a[4] = (int)rp[8];  a[5] = (int)rp[9];
    a[6] = (int)rp[12]; a[7] = (int)rp[13];
#pragma unroll
    for (int nt = 0; nt < NT; ++nt) {
      v8i b = load_bfrag(pB, ntBase + nt, numK, kc, lane);
      acc[nt] = __builtin_amdgcn_wmma_i32_16x16x64_iu8(
          true, a, true, b, acc[nt], false, false);
    }
  }
#pragma unroll
  for (int nt = 0; nt < NT; ++nt) {
    int ch = (ntBase + nt) * 16 + (lane & 15);
    if (ch >= NoutReal) continue;
    float sc = scale[ch], sf = shift[ch];
#pragma unroll
    for (int rr = 0; rr < 8; ++rr) {
      int row = mt * 16 + rr + half * 8;
      float f = (float)acc[nt][rr] * sc + sf;
      if (SOUT) outS[(long)row * NoutReal + ch] = fsign8(f);
      else      outF[row * 16 + ch] = f;
    }
  }
}


__global__ void lsm_kernel(const float* __restrict__ lg, float* __restrict__ out) {
  int r = threadIdx.x;
  if (r >= 256) return;
  const float* row = lg + r * 16;
  float mx = row[0];
  for (int i = 1; i < 10; ++i) mx = fmaxf(mx, row[i]);
  float s = 0.f;
  for (int i = 0; i < 10; ++i) s += expf(row[i] - mx);
  float lse = mx + logf(s);
  for (int i = 0; i < 10; ++i) out[r * 10 + i] = row[i] - lse;
}


static inline unsigned nblk(long n) { return (unsigned)((n + 255) / 256); }

extern "C" void kernel_launch(void* const* d_in, const int* in_sizes, int n_in,
                              void* d_out, int out_size, void* d_ws, size_t ws_size,
                              hipStream_t stream) {
  (void)in_sizes; (void)n_in; (void)out_size; (void)ws_size;
  const float* x = (const float*)d_in[0];
  const float *cw[6], *cb[6], *bg[6], *bb[6], *bm[6], *bv[6];
  for (int i = 0; i < 6; ++i) {
    int base = 1 + i * 6;
    cw[i] = (const float*)d_in[base + 0]; cb[i] = (const float*)d_in[base + 1];
    bg[i] = (const float*)d_in[base + 2]; bb[i] = (const float*)d_in[base + 3];
    bm[i] = (const float*)d_in[base + 4]; bv[i] = (const float*)d_in[base + 5];
  }
  const float* fw1 = (const float*)d_in[37]; const float* fb1 = (const float*)d_in[38];
  const float* g1  = (const float*)d_in[39]; const float* b1  = (const float*)d_in[40];
  const float* m1  = (const float*)d_in[41]; const float* v1  = (const float*)d_in[42];
  const float* fw2 = (const float*)d_in[43]; const float* fb2 = (const float*)d_in[44];
  const float* g2  = (const float*)d_in[45]; const float* b2  = (const float*)d_in[46];
  const float* m2  = (const float*)d_in[47]; const float* v2  = (const float*)d_in[48];
  const float* fw3 = (const float*)d_in[49]; const float* fb3 = (const float*)d_in[50];
  const float* m3  = (const float*)d_in[51]; const float* v3  = (const float*)d_in[52];

  char* base = (char*)d_ws; size_t off = 0;
  auto alloc = [&](size_t n) -> char* {
    char* r = base + off; off = (off + n + 255) & ~(size_t)255; return r;
  };

  const int Ci[5] = {384, 384, 768, 768, 1536};
  const int Co[5] = {384, 768, 768, 1536, 512};
  const int PL[5] = {1, 0, 1, 0, 1};
  const int CoAll[6] = {384, 384, 768, 768, 1536, 512};

  int8_t* w1s = (int8_t*)alloc(384 * 27);
  int8_t* pWc[5];
  for (int i = 0; i < 5; ++i) pWc[i] = (int8_t*)alloc((size_t)9 * Ci[i] * Co[i]);
  int8_t* pF1 = (int8_t*)alloc(8192L * 1024);
  int8_t* pF2 = (int8_t*)alloc(1024L * 1024);
  int8_t* pF3 = (int8_t*)alloc(1024L * 16);
  float *scc[6], *shc[6];
  for (int i = 0; i < 6; ++i) {
    scc[i] = (float*)alloc((size_t)CoAll[i] * 4);
    shc[i] = (float*)alloc((size_t)CoAll[i] * 4);
  }
  float* scf1 = (float*)alloc(1024 * 4); float* shf1 = (float*)alloc(1024 * 4);
  float* scf2 = (float*)alloc(1024 * 4); float* shf2 = (float*)alloc(1024 * 4);
  float* scf3 = (float*)alloc(16 * 4);   float* shf3 = (float*)alloc(16 * 4);
  int8_t* bufA  = (int8_t*)alloc(256L * 32 * 32 * 384);
  int8_t* bufB  = (int8_t*)alloc(256L * 16 * 16 * 384);
  float* logits = (float*)alloc(256 * 16 * 4);

  sign8_kernel<<<nblk(384 * 27), 256, 0, stream>>>(cw[0], w1s, 384 * 27);
  for (int i = 0; i < 5; ++i) {
    long tot = (long)9 * Ci[i] * Co[i];
    pack_conv_w<<<nblk(tot), 256, 0, stream>>>(cw[i + 1], pWc[i], Ci[i], Co[i], tot);
  }
  pack_fc_w<<<nblk(8192L * 1024), 256, 0, stream>>>(fw1, pF1, 8192, 1024, 512, 8192L * 1024);
  pack_fc_w<<<nblk(1024L * 1024), 256, 0, stream>>>(fw2, pF2, 1024, 1024, 0, 1024L * 1024);
  pack_fc_w<<<nblk(1024L * 16),   256, 0, stream>>>(fw3, pF3, 1024, 10,   0, 1024L * 16);
  for (int i = 0; i < 6; ++i)
    bn_prep<<<nblk(CoAll[i]), 256, 0, stream>>>(cb[i], bg[i], bb[i], bm[i], bv[i],
                                                scc[i], shc[i], CoAll[i], 1);
  bn_prep<<<nblk(1024), 256, 0, stream>>>(fb1, g1, b1, m1, v1, scf1, shf1, 1024, 1);
  bn_prep<<<nblk(1024), 256, 0, stream>>>(fb2, g2, b2, m2, v2, scf2, shf2, 1024, 1);
  bn_prep<<<nblk(10),   256, 0, stream>>>(fb3, fb3, fb3, m3, v3, scf3, shf3, 10, 0);

  conv1_kernel<<<nblk(256L * 32 * 32 * 384), 256, 0, stream>>>(x, w1s, scc[0], shc[0], bufA);

  {
    long TW;
    TW = (256L * 16 * 16 / 4) * (384 / 64);
    bconv_wmma<1, 32><<<(unsigned)(TW / 8), 256, 0, stream>>>(
        bufA, pWc[0], scc[1], shc[1], bufB, 384, 384, (int)TW);
    TW = (256L * 16 * 16 / 16) * (768 / 64);
    bconv_wmma<0, 16><<<(unsigned)(TW / 8), 256, 0, stream>>>(
        bufB, pWc[1], scc[2], shc[2], bufA, 384, 768, (int)TW);
    TW = (256L * 8 * 8 / 4) * (768 / 64);
    bconv_wmma<1, 16><<<(unsigned)(TW / 8), 256, 0, stream>>>(
        bufA, pWc[2], scc[3], shc[3], bufB, 768, 768, (int)TW);
    TW = (256L * 8 * 8 / 16) * (1536 / 64);
    bconv_wmma<0, 8><<<(unsigned)(TW / 8), 256, 0, stream>>>(
        bufB, pWc[3], scc[4], shc[4], bufA, 768, 1536, (int)TW);
    TW = (256L * 4 * 4 / 4) * (512 / 64);
    bconv_wmma<1, 8><<<(unsigned)(TW / 8), 256, 0, stream>>>(
        bufA, pWc[4], scc[5], shc[5], bufB, 1536, 512, (int)TW);
  }

  bfc_wmma<4, 1><<<32, 256, 0, stream>>>(
      bufB, pF1, scf1, shf1, bufA, nullptr, 8192, 1024, 256);
  bfc_wmma<4, 1><<<32, 256, 0, stream>>>(
      bufA, pF2, scf2, shf2, bufB, nullptr, 1024, 1024, 256);
  bfc_wmma<1, 0><<<2, 256, 0, stream>>>(
      bufB, pF3, scf3, shf3, nullptr, logits, 1024, 10, 16);

  lsm_kernel<<<1, 256, 0, stream>>>(logits, (float*)d_out);
}
